// VectorQuantizer_498216206954
// MI455X (gfx1250) — matrix operand order repaired and checked by reading, not compiled, not run on hardware
//
#include <hip/hip_runtime.h>
#include <stdint.h>

// ---------------- problem constants ----------------
#define N_ROWS   32768      // 4*8*32*32
#define DIM      256
#define K_CODES  8192
#define BM       64         // rows per block
#define BN       32         // codebook cols per chunk
#define NCHUNKS  (K_CODES / BN)   // 256
#define BROW     (DIM + 8)  // bf16 LDS row stride: 528B (16B aligned, bank stride 4)
#define ZROW     (DIM + 4)  // fp32 staging row stride
#define TILE_B   (BN * BROW * 2)          // one bf16 buffer: 16896 B
#define BUF_B    (2 * TILE_B)             // hi+lo: 33792 B
#define LDS_DYN  (2 * BUF_B)              // double buffered: 67584 B

typedef __bf16        bf16x16 __attribute__((ext_vector_type(16)));
typedef float         f32x8   __attribute__((ext_vector_type(8)));
typedef float         f32x4   __attribute__((ext_vector_type(4)));
typedef unsigned int  u32x4   __attribute__((ext_vector_type(4)));
typedef unsigned int  u32x2   __attribute__((ext_vector_type(2)));
typedef int           i32x4g  __attribute__((vector_size(16)));  // matches builtin param

union Frag { bf16x16 v; unsigned short u[16]; u32x4 q[2]; };

__device__ __forceinline__ unsigned short f32_to_bf16_rne(float f) {
  unsigned int u = __float_as_uint(f);
  u += 0x7FFFu + ((u >> 16) & 1u);       // round to nearest even
  return (unsigned short)(u >> 16);
}
__device__ __forceinline__ float bf16_bits_to_f32(unsigned short h) {
  return __uint_as_float(((unsigned int)h) << 16);
}

// ---- CDNA5 async global->LDS path (guarded; fallback = sync vector copy) ----
#if __has_builtin(__builtin_amdgcn_global_load_async_to_lds_b128)
#define HAVE_ASYNC 1
#define ASYNC_B128(gsrc, ldst)                                               \
  __builtin_amdgcn_global_load_async_to_lds_b128(                            \
      (__attribute__((address_space(1))) i32x4g*)(i32x4g*)(void*)(gsrc),     \
      (__attribute__((address_space(3))) i32x4g*)(i32x4g*)(void*)(ldst),     \
      0, 0)
#else
#define HAVE_ASYNC 0
#define ASYNC_B128(gsrc, ldst)                                               \
  (*(u32x4*)(ldst) = *(const u32x4*)(gsrc))
#endif

#if __has_builtin(__builtin_amdgcn_s_wait_asynccnt)
#define WAIT_ASYNC(n) __builtin_amdgcn_s_wait_asynccnt(n)
#elif HAVE_ASYNC
#define WAIT_ASYNC(n) asm volatile("s_wait_asynccnt %0" ::"i"(n))
#else
#define WAIT_ASYNC(n) ((void)0)
#endif

// ---------------- kernel 0: codebook -> bf16 hi/lo + norms (runs once/call) --
__global__ __launch_bounds__(256) void vq_prep_kernel(
    const float* __restrict__ cb,
    unsigned short* __restrict__ cb_hi, unsigned short* __restrict__ cb_lo,
    float* __restrict__ cnorm)
{
  const int t   = threadIdx.x;
  const int row = blockIdx.x * 32 + (t >> 3);
  const int seg = t & 7;
  const float* cp = cb + (long)row * DIM + seg * 32;
  unsigned short* hp = cb_hi + (long)row * DIM + seg * 32;
  unsigned short* lp = cb_lo + (long)row * DIM + seg * 32;
  float ss = 0.f;
  #pragma unroll
  for (int i = 0; i < 32; i += 4) {
    const f32x4 v4 = *(const f32x4*)(cp + i);
    unsigned short hb[4], lb[4];
    #pragma unroll
    for (int e = 0; e < 4; ++e) {
      const float v = v4[e];
      hb[e] = f32_to_bf16_rne(v);
      lb[e] = f32_to_bf16_rne(v - bf16_bits_to_f32(hb[e]));
      ss += v * v;
    }
    u32x2 ph, pl;
    ph[0] = (unsigned)hb[0] | ((unsigned)hb[1] << 16);
    ph[1] = (unsigned)hb[2] | ((unsigned)hb[3] << 16);
    pl[0] = (unsigned)lb[0] | ((unsigned)lb[1] << 16);
    pl[1] = (unsigned)lb[2] | ((unsigned)lb[3] << 16);
    *(u32x2*)(hp + i) = ph;
    *(u32x2*)(lp + i) = pl;
  }
  ss += __shfl_xor(ss, 1);
  ss += __shfl_xor(ss, 2);
  ss += __shfl_xor(ss, 4);
  if (seg == 0) cnorm[row] = ss;
}

// ---------------- main fused distance + argmin kernel ------------------------
__global__ __launch_bounds__(256) void vq_argmin_kernel(
    const float* __restrict__ z, const float* __restrict__ cb,
    const unsigned short* __restrict__ cb_hi,
    const unsigned short* __restrict__ cb_lo,
    const float* __restrict__ cnormg,
    float* __restrict__ zq_out, float* __restrict__ idx_out,
    float* __restrict__ loss_ws)
{
  extern __shared__ __align__(16) unsigned char smem_raw[];  // LDS_DYN bytes
  __shared__ float s_znorm[BM];
  __shared__ float s_redD[2][BM];
  __shared__ int   s_redI[2][BM];
  __shared__ int   s_bestI[BM];
  __shared__ float s_wls[8];

  const int t     = threadIdx.x;
  const int lane  = t & 31;
  const int wave  = t >> 5;
  const int waveM = wave >> 1;     // 0..3 -> 16-row group
  const int waveN = wave & 1;      // 0..1 -> 16-col group
  const int h     = lane >> 4;     // lane half
  const int nl    = lane & 15;
  const int rbase = waveM * 16;
  const int nbase = waveN * 16;
  const long blockRow = (long)blockIdx.x * BM;

  float* zstage = (float*)smem_raw;     // phase A reuse of tile memory

  // ---- Phase A: stage z rows (2 passes of 32), norms, build A frags in regs ----
  Frag a_hi[8], a_lo[8];
  #pragma unroll
  for (int p = 0; p < 2; ++p) {
    __syncthreads();
    {
      const int rloc = t >> 3;     // 0..31
      const int seg  = t & 7;      // 32 floats each
      const float* zp = z + (blockRow + 32 * p + rloc) * (long)DIM + seg * 32;
      float* zs = zstage + rloc * ZROW + seg * 32;
      float ss = 0.f;
      #pragma unroll
      for (int i = 0; i < 32; i += 4) {
        f32x4 v4 = *(const f32x4*)(zp + i);
        *(f32x4*)(zs + i) = v4;
        ss += v4[0]*v4[0] + v4[1]*v4[1] + v4[2]*v4[2] + v4[3]*v4[3];
      }
      ss += __shfl_xor(ss, 1);
      ss += __shfl_xor(ss, 2);
      ss += __shfl_xor(ss, 4);
      if (seg == 0) s_znorm[32 * p + rloc] = ss;
    }
    __syncthreads();
    if ((waveM >> 1) == p) {       // waves owning rows of this pass
      const int rloc = 16 * (waveM & 1) + nl;
      #pragma unroll
      for (int s = 0; s < 8; ++s) {
        #pragma unroll
        for (int j = 0; j < 16; ++j) {
          // A 16x32 layout (contiguous, matches B): half0 K=0..15 ; half1 K=16..31
          const int K = 16 * h + j;
          const float v = zstage[rloc * ZROW + s * 32 + K];
          const unsigned short hb = f32_to_bf16_rne(v);
          a_hi[s].u[j] = hb;
          a_lo[s].u[j] = f32_to_bf16_rne(v - bf16_bits_to_f32(hb));
        }
      }
    }
  }
  __syncthreads();   // all A builds done before tile memory is reused

  float bestD[8];
  int   bestI[8];
  float zn[8];
  #pragma unroll
  for (int v = 0; v < 8; ++v) {
    bestD[v] = 3.4e38f; bestI[v] = 0;
    zn[v] = s_znorm[rbase + v + 8 * h];  // C-layout: row M = v + 8*h
  }

  // per-thread segment descriptors for the tile copy: 4 x 16B (hi) + 4 x (lo)
  // seg = t + 256*q ; col = seg>>5 ; 16B sub-offset = (seg&31)*16
  const unsigned char* ghB = (const unsigned char*)cb_hi;  // 512 B per row
  const unsigned char* glB = (const unsigned char*)cb_lo;

  auto issue_chunk = [&](int chunk, int buf) {
    unsigned char* lbase = smem_raw + buf * BUF_B;
    const long gbase = (long)chunk * (BN * DIM * 2);  // bytes into cb_hi/cb_lo
    #pragma unroll
    for (int q = 0; q < 4; ++q) {
      const int seg = t + 256 * q;
      const int col = seg >> 5;
      const int off = (seg & 31) * 16;
      unsigned char* ld = lbase + col * (BROW * 2) + off;
      const long     gd = gbase + (long)col * (DIM * 2) + off;
      ASYNC_B128(ghB + gd, ld);
      ASYNC_B128(glB + gd, ld + TILE_B);
    }
  };

  issue_chunk(0, 0);
  issue_chunk(1, 1);

  // ---- Main loop over codebook chunks (double-buffered async staging) ----
  for (int chunk = 0; chunk < NCHUNKS; ++chunk) {
    const int bufA = chunk & 1;
    if (chunk == NCHUNKS - 1) { WAIT_ASYNC(0); } else { WAIT_ASYNC(8); }
    __syncthreads();   // current buffer complete across all waves

    unsigned short* sB_hi = (unsigned short*)(smem_raw + bufA * BUF_B);
    unsigned short* sB_lo = sB_hi + BN * BROW;

    const int   colG = chunk * BN + nbase + nl;
    const float cn   = cnormg[colG];           // precomputed ||e||^2 (L2-resident)

    // WMMA: dot(z,e) for this wave's 16x16 tile, bf16 hi/lo compensated
    f32x8 acc = {0.f, 0.f, 0.f, 0.f, 0.f, 0.f, 0.f, 0.f};
    const int colB = (nbase + nl) * BROW;      // B layout: lane = N, half = K-half
    #pragma unroll
    for (int s = 0; s < 8; ++s) {
      Frag fbh, fbl;
      const u32x4* ph = (const u32x4*)(sB_hi + colB + s * 32 + 16 * h);
      const u32x4* pl = (const u32x4*)(sB_lo + colB + s * 32 + 16 * h);
      fbh.q[0] = ph[0]; fbh.q[1] = ph[1];
      fbl.q[0] = pl[0]; fbl.q[1] = pl[1];
      acc = __builtin_amdgcn_wmma_f32_16x16x32_bf16(false, a_hi[s].v, false, fbh.v,
                                                    (short)0, acc, false, false);
      acc = __builtin_amdgcn_wmma_f32_16x16x32_bf16(false, a_lo[s].v, false, fbh.v,
                                                    (short)0, acc, false, false);
      acc = __builtin_amdgcn_wmma_f32_16x16x32_bf16(false, a_hi[s].v, false, fbl.v,
                                                    (short)0, acc, false, false);
    }

    // distances + running argmin
    #pragma unroll
    for (int v = 0; v < 8; ++v) {
      const float dist = zn[v] + cn - 2.0f * acc[v];
      if (dist < bestD[v]) { bestD[v] = dist; bestI[v] = colG; }
    }

    __syncthreads();   // all waves done reading bufA before it is refilled
    if (chunk + 2 < NCHUNKS) issue_chunk(chunk + 2, bufA);
  }

  // ---- argmin reduction: across 16 lanes, then across the 2 N-waves ----
  #pragma unroll
  for (int v = 0; v < 8; ++v) {
    float d = bestD[v]; int bi = bestI[v];
    #pragma unroll
    for (int off = 8; off >= 1; off >>= 1) {
      const float od = __shfl_xor(d, off);
      const int   oi = __shfl_xor(bi, off);
      if (od < d || (od == d && oi < bi)) { d = od; bi = oi; }
    }
    if (nl == 0) {
      const int row = rbase + v + 8 * h;
      s_redD[waveN][row] = d;
      s_redI[waveN][row] = bi;
    }
  }
  __syncthreads();
  if (t < BM) {
    const float d0 = s_redD[0][t]; const int i0 = s_redI[0][t];
    const float d1 = s_redD[1][t]; const int i1 = s_redI[1][t];
    const int bi = (d1 < d0 || (d1 == d0 && i1 < i0)) ? i1 : i0;
    s_bestI[t] = bi;
    idx_out[blockRow + t] = (float)bi;   // indices as float in concatenated out
  }
  __syncthreads();

  // ---- gather z_q (fp32 codebook), write out, local loss ----
  float ls = 0.f;
  {
    const int row = t >> 2;          // 4 threads/row, 64 floats each
    const int seg = t & 3;
    const int bi  = s_bestI[row];
    const float* cr = cb + (long)bi * DIM + seg * 64;
    const float* zr = z + (blockRow + row) * (long)DIM + seg * 64;
    float* orow = zq_out + (blockRow + row) * (long)DIM + seg * 64;
    #pragma unroll
    for (int i = 0; i < 64; i += 4) {
      const f32x4 c4 = *(const f32x4*)(cr + i);
      const f32x4 z4 = *(const f32x4*)(zr + i);
      *(f32x4*)(orow + i) = c4;
      const f32x4 dd = c4 - z4;
      ls += dd[0]*dd[0] + dd[1]*dd[1] + dd[2]*dd[2] + dd[3]*dd[3];
    }
  }
  // deterministic block loss: shuffle-reduce per wave, fixed-order sum
  ls += __shfl_xor(ls, 1);
  ls += __shfl_xor(ls, 2);
  ls += __shfl_xor(ls, 4);
  ls += __shfl_xor(ls, 8);
  ls += __shfl_xor(ls, 16);
  if (lane == 0) s_wls[wave] = ls;
  __syncthreads();
  if (t == 0) {
    float tot = 0.f;
    #pragma unroll
    for (int w = 0; w < 8; ++w) tot += s_wls[w];
    loss_ws[blockIdx.x] = tot;
  }
}

__global__ __launch_bounds__(256) void vq_loss_reduce_kernel(
    const float* __restrict__ ws, float* __restrict__ loss_out)
{
  __shared__ float sm[256];
  const int t = threadIdx.x;
  sm[t] = ws[t] + ws[t + 256];
  __syncthreads();
  for (int s = 128; s > 0; s >>= 1) {
    if (t < s) sm[t] += sm[t + s];
    __syncthreads();
  }
  if (t == 0)
    loss_out[0] = sm[0] * (1.25f / ((float)N_ROWS * (float)DIM));
}

extern "C" void kernel_launch(void* const* d_in, const int* in_sizes, int n_in,
                              void* d_out, int out_size, void* d_ws, size_t ws_size,
                              hipStream_t stream) {
  const float* z  = (const float*)d_in[0];   // [4,8,32,32,256] fp32
  const float* cb = (const float*)d_in[1];   // [8192,256] fp32
  float* out = (float*)d_out;
  float* zq_out   = out;                                 // 8,388,608 floats
  float* loss_out = out + (size_t)N_ROWS * DIM;          // 1 float
  float* idx_out  = loss_out + 1;                        // 32768 floats

  // workspace layout (needs ~8.43 MB)
  unsigned char* ws = (unsigned char*)d_ws;
  unsigned short* cb_hi = (unsigned short*)(ws);                        // 4 MB
  unsigned short* cb_lo = (unsigned short*)(ws + (size_t)4194304);      // 4 MB
  float*          cnorm = (float*)(ws + (size_t)8388608);               // 32 KB
  float*          lossp = (float*)(ws + (size_t)8421376);               // 2 KB

  vq_prep_kernel<<<dim3(K_CODES / 32), dim3(256), 0, stream>>>(cb, cb_hi, cb_lo, cnorm);

  const int nblocks = N_ROWS / BM;  // 512
  vq_argmin_kernel<<<dim3(nblocks), dim3(256), LDS_DYN, stream>>>(
      z, cb, cb_hi, cb_lo, cnorm, zq_out, idx_out, lossp);

  vq_loss_reduce_kernel<<<dim3(1), dim3(256), 0, stream>>>(lossp, loss_out);
}
